// LstmFusion_59012850647318
// MI455X (gfx1250) — matrix operand order repaired and checked by reading, not compiled, not run on hardware
//
#include <hip/hip_runtime.h>
#include <hip/hip_bf16.h>

typedef __bf16 bf16_t;
typedef __attribute__((ext_vector_type(8)))  bf16_t v8bf;
typedef __attribute__((ext_vector_type(16))) bf16_t v16bf;
typedef __attribute__((ext_vector_type(8)))  float  v8f;
typedef __attribute__((ext_vector_type(4)))  int    v4i;

#define B_   64
#define L_   1024
#define IN_  512
#define H_   512
#define G_   1536          // 3*H
#define NC_  16            // h-columns per scan block
#define NBLK_SCAN 32       // blocks per direction in scan (H_/NC_)

// ---------------- workspace layout (bytes) ----------------
// bar      : int [2][L]                     @ 0        (8192 B)
// hf       : f32 [2][B][H]                  @ 8192     (262144 B)
// hb       : bf16[2 parity][2 dir][B][H]    @ 270336   (262144 B)
// wih_bf   : bf16[2][G][IN]                 @ 532480   (3145728 B)
// whh_bf   : bf16[2][G][H]                  @ 3678208  (3145728 B)
// feats_bf : bf16[B][L][IN]                 @ 6823936  (67108864 B)
// gx       : bf16[2][L][B][G]               @ 73932800 (402653184 B)
#define WS_BAR    0
#define WS_HF     8192
#define WS_HB     270336
#define WS_WIH    532480
#define WS_WHH    3678208
#define WS_FEATS  6823936
#define WS_GX     73932800
#define WS_ZERO_BYTES 532480   // bar + hf + hb must be zeroed every call

// ---------- gfx1250 async memory->LDS path (probe via __has_builtin) --------
#if defined(__AMDGCN__) && __has_builtin(__builtin_amdgcn_global_load_async_to_lds_b128) && __has_builtin(__builtin_amdgcn_s_wait_asynccnt)
#define HAVE_ASYNC_LDS 1
#else
#define HAVE_ASYNC_LDS 0
#endif

__device__ __forceinline__ void stage16(const bf16_t* g, bf16_t* l) {
#if HAVE_ASYNC_LDS
  // direct global->LDS DMA (no VGPR round trip), tracked by ASYNCcnt.
  // prototype: (v4i AS1* gsrc, v4i AS3* ldst, imm offset, imm cpol)
  __builtin_amdgcn_global_load_async_to_lds_b128(
      (__attribute__((address_space(1))) v4i*)g,
      (__attribute__((address_space(3))) v4i*)l, 0, 0);
#else
  *(v8bf*)l = *(const v8bf*)g;
#endif
}

__device__ __forceinline__ void stage_wait() {
#if HAVE_ASYNC_LDS
  __builtin_amdgcn_s_wait_asynccnt(0);
#endif
}

__device__ __forceinline__ v16bf cat16(v8bf lo, v8bf hi) {
  return __builtin_shufflevector(lo, hi, 0,1,2,3,4,5,6,7,8,9,10,11,12,13,14,15);
}

// ---------------- f32 -> bf16 conversion ----------------
__global__ void cvt_f32_bf16(const float* __restrict__ src,
                             bf16_t* __restrict__ dst, int n) {
  int i = blockIdx.x * blockDim.x + threadIdx.x;
  int stride = gridDim.x * blockDim.x;
  for (; i < n; i += stride) dst[i] = (bf16_t)src[i];
}

// ---------------- input projection: gx = x @ w_ih^T + b_ih ----------------
// grid (512, 24, 2): block = 128 M-rows x 64 N-cols, 256 threads = 8 waves.
// Wave wv: M-tile (16 rows), 4 N-tiles (64 cols), K=512 in 16 steps of 32.
__global__ __launch_bounds__(256, 1)
void gru_inproj(const bf16_t* __restrict__ featsb,   // [B*L, IN] bf16
                const bf16_t* __restrict__ wihb,     // [2][G][IN] bf16
                const float*  __restrict__ bih_lr,
                const float*  __restrict__ bih_rl,
                bf16_t* __restrict__ gx)             // [2][L][B][G] bf16
{
  extern __shared__ char smem_raw[];
  bf16_t* wlds = (bf16_t*)smem_raw;                  // [64][512] = 64 KB

  const int d     = blockIdx.z;
  const int mBase = blockIdx.x * 128;
  const int nBase = blockIdx.y * 64;
  const int tid   = threadIdx.x;

  // stage weight slice rows nBase..nBase+63 into LDS (bf16)
  const bf16_t* wsrc = wihb + ((size_t)d * G_ + nBase) * IN_;
  for (int c = tid; c < (64 * 512) / 8; c += 256) {
    int row = c >> 6;              // 64 chunks of 8 bf16 per row
    int off = (c & 63) * 8;
    stage16(wsrc + (size_t)row * IN_ + off, wlds + row * 512 + off);
  }
  stage_wait();
  __syncthreads();

  const int wv   = tid >> 5;
  const int lane = tid & 31;
  const int l15  = lane & 15;
  const int hi   = lane >> 4;                         // lane half
  const int m0   = mBase + 16 * wv;
  const bf16_t* arow = featsb + (size_t)(m0 + l15) * IN_;

  v8f acc[4];
#pragma unroll
  for (int i = 0; i < 4; ++i)
#pragma unroll
    for (int e = 0; e < 8; ++e) acc[i][e] = 0.0f;

  for (int kk = 0; kk < 16; ++kk) {
    const int k0 = kk * 32;
    v8bf alo = *(const v8bf*)(arow + k0 + hi * 8);
    v8bf ahi = *(const v8bf*)(arow + k0 + 16 + hi * 8);
    v16bf afrag = cat16(alo, ahi);
#pragma unroll
    for (int nt = 0; nt < 4; ++nt) {
      const bf16_t* bp = wlds + (16 * nt + l15) * 512 + k0 + hi * 8;
      v16bf bfrag = cat16(*(const v8bf*)bp, *(const v8bf*)(bp + 16));
      acc[nt] = __builtin_amdgcn_wmma_f32_16x16x32_bf16(
          false, afrag, false, bfrag, (short)0, acc[nt], false, false);
    }
  }

  const float* bih = d ? bih_rl : bih_lr;
  bf16_t* gxd = gx + (size_t)d * L_ * B_ * G_;
#pragma unroll
  for (int nt = 0; nt < 4; ++nt) {
    const int g    = nBase + 16 * nt + l15;
    const float bs = bih[g];
#pragma unroll
    for (int r = 0; r < 8; ++r) {
      int m = m0 + hi * 8 + r;
      int b = m >> 10;                 // m / L
      int l = m & 1023;                // m % L
      int t = d ? (L_ - 1 - l) : l;    // time-major, reversed for rl
      gxd[((size_t)t * B_ + b) * G_ + g] = (bf16_t)(acc[nt][r] + bs);
    }
  }
}

// ---------------- recurrent scan ----------------
// grid (32, 2): block owns NC_=16 h-columns of one direction.
// 128 threads = 4 waves; wave wv = batch rows 16*wv..16*wv+15.
// Per step: gh tile = h(bf16)[64,512] x w_hh_slice[512,48] via 48 WMMAs,
// fused GRU gates, per-step device-scope barrier across the 32 blocks.
__global__ __launch_bounds__(128, 1)
void gru_scan(const bf16_t* __restrict__ whhb,   // [2][G][H] bf16
              const bf16_t* __restrict__ gx,     // [2][L][B][G] bf16
              const float*  __restrict__ mask,   // [B][L]
              const float*  __restrict__ bhh_lr,
              const float*  __restrict__ bhh_rl,
              float*  __restrict__ hf,           // [2][B][H] f32 master
              bf16_t* __restrict__ hb,           // [2 par][2 dir][B][H] bf16
              int*    __restrict__ bar,          // [2][L]
              float*  __restrict__ out)          // [B][L][2H]
{
  extern __shared__ char smem_raw[];
  bf16_t* wlds = (bf16_t*)smem_raw;              // [48][512] = 48 KB

  const int d   = blockIdx.y;
  const int c0  = blockIdx.x * NC_;
  const int tid = threadIdx.x;
  const int wv = tid >> 5, lane = tid & 31, l15 = lane & 15, hi = lane >> 4;
  const int m0 = 16 * wv;

  // stage w_hh rows {gate*512 + c0 + jj : gate<3, jj<16} once (reused 1024 steps)
  for (int c = tid; c < (48 * 512) / 8; c += 128) {
    int lg = c >> 6, off = (c & 63) * 8;
    int gate = lg >> 4, jj = lg & 15;
    stage16(whhb + ((size_t)d * G_ + gate * H_ + c0 + jj) * H_ + off,
            wlds + lg * 512 + off);
  }
  stage_wait();
  __syncthreads();

  const float* bhh = d ? bhh_rl : bhh_lr;
  const int jlane = c0 + l15;
  const float bh_r = bhh[jlane];
  const float bh_z = bhh[512 + jlane];
  const float bh_n = bhh[1024 + jlane];

  for (int t = 0; t < L_; ++t) {
    const bf16_t* hread = hb + (size_t)((t & 1) * 2 + d) * (B_ * H_);
    bf16_t* hwrite      = hb + (size_t)(((t + 1) & 1) * 2 + d) * (B_ * H_);
    const bf16_t* gxt   = gx + ((size_t)d * L_ + t) * (size_t)(B_ * G_);

    // prefetch next step's gx rows for this lane (global_prefetch_b8)
    if (t + 1 < L_) {
      const bf16_t* gxn = gxt + (size_t)(B_ * G_);
      size_t o = (size_t)(m0 + hi * 8) * G_ + jlane;
      __builtin_prefetch(gxn + o, 0, 1);
      __builtin_prefetch(gxn + o + 512, 0, 1);
      __builtin_prefetch(gxn + o + 1024, 0, 1);
    }

    v8f acc[3];
#pragma unroll
    for (int i = 0; i < 3; ++i)
#pragma unroll
      for (int e = 0; e < 8; ++e) acc[i][e] = 0.0f;

    const bf16_t* arow = hread + (size_t)(m0 + l15) * H_;
    for (int kk = 0; kk < 16; ++kk) {
      const int k0 = kk * 32;
      v8bf alo = *(const v8bf*)(arow + k0 + hi * 8);
      v8bf ahi = *(const v8bf*)(arow + k0 + 16 + hi * 8);
      v16bf afrag = cat16(alo, ahi);
#pragma unroll
      for (int g3 = 0; g3 < 3; ++g3) {
        const bf16_t* bp = wlds + (16 * g3 + l15) * 512 + k0 + hi * 8;
        v16bf bfrag = cat16(*(const v8bf*)bp, *(const v8bf*)(bp + 16));
        acc[g3] = __builtin_amdgcn_wmma_f32_16x16x32_bf16(
            false, afrag, false, bfrag, (short)0, acc[g3], false, false);
      }
    }

    const int l_eff = d ? (L_ - 1 - t) : t;
#pragma unroll
    for (int r = 0; r < 8; ++r) {
      const int b = m0 + hi * 8 + r;
      float xr = (float)gxt[(size_t)b * G_ + jlane];
      float xz = (float)gxt[(size_t)b * G_ + 512 + jlane];
      float xn = (float)gxt[(size_t)b * G_ + 1024 + jlane];
      float hold = hf[((size_t)d * B_ + b) * H_ + jlane];
      float hrv = acc[0][r] + bh_r;
      float hzv = acc[1][r] + bh_z;
      float hnv = acc[2][r] + bh_n;
      float rg = 1.0f / (1.0f + __expf(-(xr + hrv)));
      float zg = 1.0f / (1.0f + __expf(-(xz + hzv)));
      float ng = tanhf(xn + rg * hnv);
      float hnew = (1.0f - zg) * ng + zg * hold;
      float mk = mask[(size_t)b * L_ + l_eff];
      float hv = mk * hnew + (1.0f - mk) * hold;
      hf[((size_t)d * B_ + b) * H_ + jlane] = hv;
      hwrite[(size_t)b * H_ + jlane] = (bf16_t)hv;
      out[((size_t)b * L_ + l_eff) * (2 * H_) + d * H_ + jlane] = hv;
    }

    // release h writes, then per-step barrier across this direction's blocks
    __threadfence();
    __syncthreads();
    if (tid == 0) {
      __hip_atomic_fetch_add(&bar[d * L_ + t], 1, __ATOMIC_ACQ_REL,
                             __HIP_MEMORY_SCOPE_AGENT);
      while (__hip_atomic_load(&bar[d * L_ + t], __ATOMIC_ACQUIRE,
                               __HIP_MEMORY_SCOPE_AGENT) < NBLK_SCAN) {
        __builtin_amdgcn_s_sleep(1);
      }
    }
    __syncthreads();
  }
}

extern "C" void kernel_launch(void* const* d_in, const int* in_sizes, int n_in,
                              void* d_out, int out_size, void* d_ws, size_t ws_size,
                              hipStream_t stream) {
  const float* feats  = (const float*)d_in[0];
  const float* mask   = (const float*)d_in[1];
  const float* wih_lr = (const float*)d_in[2];
  const float* whh_lr = (const float*)d_in[3];
  const float* bih_lr = (const float*)d_in[4];
  const float* bhh_lr = (const float*)d_in[5];
  const float* wih_rl = (const float*)d_in[6];
  const float* whh_rl = (const float*)d_in[7];
  const float* bih_rl = (const float*)d_in[8];
  const float* bhh_rl = (const float*)d_in[9];
  float* out = (float*)d_out;

  char* ws = (char*)d_ws;
  int*    bar    = (int*)(ws + WS_BAR);
  float*  hf     = (float*)(ws + WS_HF);
  bf16_t* hb     = (bf16_t*)(ws + WS_HB);
  bf16_t* wihb   = (bf16_t*)(ws + WS_WIH);
  bf16_t* whhb   = (bf16_t*)(ws + WS_WHH);
  bf16_t* featsb = (bf16_t*)(ws + WS_FEATS);
  bf16_t* gxbuf  = (bf16_t*)(ws + WS_GX);

  // zero barrier counters + h state every call (deterministic, capture-safe)
  (void)hipMemsetAsync(ws, 0, WS_ZERO_BYTES, stream);

  // f32 -> bf16 staging
  cvt_f32_bf16<<<2048, 256, 0, stream>>>(feats, featsb, B_ * L_ * IN_);
  cvt_f32_bf16<<<768, 256, 0, stream>>>(wih_lr, wihb, G_ * IN_);
  cvt_f32_bf16<<<768, 256, 0, stream>>>(wih_rl, wihb + (size_t)G_ * IN_, G_ * IN_);
  cvt_f32_bf16<<<768, 256, 0, stream>>>(whh_lr, whhb, G_ * H_);
  cvt_f32_bf16<<<768, 256, 0, stream>>>(whh_rl, whhb + (size_t)G_ * H_, G_ * H_);

  // input projection (both directions), time-major gx
  gru_inproj<<<dim3(512, 24, 2), 256, 65536, stream>>>(
      featsb, wihb, bih_lr, bih_rl, gxbuf);

  // sequential bidirectional scan
  gru_scan<<<dim3(NBLK_SCAN, 2, 1), 128, 49152, stream>>>(
      whhb, gxbuf, mask, bhh_lr, bhh_rl, hf, hb, bar, out);
}
